// BoundaryAwareEncoder_6674379178167
// MI455X (gfx1250) — matrix operand order repaired and checked by reading, not compiled, not run on hardware
//
#include <hip/hip_runtime.h>
#include <math.h>

// ---------------- CDNA5 WMMA types ----------------
typedef __attribute__((ext_vector_type(16))) __bf16 v16bf;
typedef __attribute__((ext_vector_type(8)))  float  v8f;
typedef __attribute__((ext_vector_type(4)))  unsigned int uint4v;

union Frag16 {
    unsigned short u[16];
    uint4v         q[2];
    v16bf          v;
};

__device__ __forceinline__ unsigned short f2bf(float f) {
    unsigned u = __float_as_uint(f);
    unsigned r = u + 0x7FFFu + ((u >> 16) & 1u);
    return (unsigned short)(r >> 16);
}
__device__ __forceinline__ float bf2f(unsigned short s) {
    return __uint_as_float(((unsigned)s) << 16);
}

// ---------------- generalized batched bf16 WMMA GEMM ----------------
// C[z] = act(A[z] @ B[z] + bias) (+ res[z])   A: MxK bf16, B: KxN bf16
// batch index z decomposes: zo = z / batchDiv, zi = z % batchDiv
struct GemmArgs {
    const unsigned short* A;
    const unsigned short* B;
    float*          Cf;
    unsigned short* Cb;
    const float* bias;
    const float* res;
    long aOuter, aInner, ldaM;            // A element (m,k) at aOff + m*ldaM + k
    long bOuter, bInner, ldbK, ldbN;      // B element (k,n) at bOff + k*ldbK + n*ldbN
    long cOuter, cInner, ldcM;            // C element (m,n) at cOff + m*ldcM + n
    long rOuter, rInner, ldrM;
    int M, N, K, act, batchDiv, tilesM32; // act: 0 none, 1 gelu(exact), 2 relu
};

__device__ __forceinline__ v8f wmma_step(const Frag16& fa, const Frag16& fb, v8f acc) {
    return __builtin_amdgcn_wmma_f32_16x16x32_bf16(
        false, fa.v, false, fb.v, (short)0, acc, false, false);
}

// Unconditional load from clamped (in-bounds) address, then bitwise mask.
// The loaded value always feeds an AND, so the compiler cannot sink the load
// under divergent control flow (no saveexec/branch per element).
__device__ __forceinline__ unsigned short mload(const unsigned short* p, bool ok) {
    unsigned short v = *p;
    unsigned short msk = ok ? (unsigned short)0xFFFFu : (unsigned short)0u;
    return (unsigned short)(v & msk);
}

__global__ void k_gemm(GemmArgs g) {
    const int lane = threadIdx.x & 31;
    const int wid  = threadIdx.x >> 5;
    const int z  = blockIdx.y;
    const int zo = z / g.batchDiv;
    const int zi = z % g.batchDiv;

    const unsigned short* A  = g.A + (long)zo * g.aOuter + (long)zi * g.aInner;
    const unsigned short* Bm = g.B + (long)zo * g.bOuter + (long)zi * g.bInner;

    const int bm = blockIdx.x % g.tilesM32;
    const int bn = blockIdx.x / g.tilesM32;
    const int m0 = bm * 32 + (wid & 1) * 16;
    const int n0 = bn * 32 + (wid >> 1) * 16;
    if (m0 >= g.M || n0 >= g.N) return;   // wave-uniform exit (EXEC stays full for wmma)

    v8f acc = {};
    const int  mA = m0 + (lane & 15);          // A: lanes hold M rows
    const int  kb = (lane >> 4) << 3;          // 0 or 8 depending on lane half
    const bool fullM = (m0 + 16 <= g.M);
    const bool fullN = (n0 + 16 <= g.N);
    const int  Kv = g.K & ~31;                 // full 32-wide K chunks

    if (fullM && fullN) {
        const unsigned short* ap = A + (long)mA * g.ldaM + kb;
        if (g.ldbN == 1) {
            // -------- hot path: contiguous B (weights / V) --------
            const unsigned short* bp = Bm + (long)kb * g.ldbK + n0 + (lane & 15);
            const long bstep = 32L * g.ldbK;
            for (int k0 = 0; k0 < Kv; k0 += 32) {
                Frag16 fa, fb;
                fa.q[0] = *reinterpret_cast<const uint4v*>(ap);
                fa.q[1] = *reinterpret_cast<const uint4v*>(ap + 16);
                #pragma unroll
                for (int e = 0; e < 16; e++) fb.u[e] = bp[(long)((e < 8) ? e : e + 8) * g.ldbK];
                __builtin_prefetch(ap + 128, 0, 1);   // global_prefetch_b8
                acc = wmma_step(fa, fb, acc);
                ap += 32; bp += bstep;
            }
        } else {
            // -------- hot path: strided B (K^T for attention scores) --------
            const unsigned short* bp = Bm + (long)kb * g.ldbK + (long)(n0 + (lane & 15)) * g.ldbN;
            const long bstep = 32L * g.ldbK;
            for (int k0 = 0; k0 < Kv; k0 += 32) {
                Frag16 fa, fb;
                fa.q[0] = *reinterpret_cast<const uint4v*>(ap);
                fa.q[1] = *reinterpret_cast<const uint4v*>(ap + 16);
                #pragma unroll
                for (int e = 0; e < 16; e++) fb.u[e] = bp[(long)((e < 8) ? e : e + 8) * g.ldbK];
                __builtin_prefetch(ap + 128, 0, 1);
                acc = wmma_step(fa, fb, acc);
                ap += 32; bp += bstep;
            }
        }
        if (Kv < g.K) {
            // K tail (e.g. dh = 80): branchless masked loads
            const int k0 = Kv;
            const unsigned short* arow = A + (long)mA * g.ldaM;
            Frag16 fa, fb;
            #pragma unroll
            for (int e = 0; e < 16; e++) {
                int k  = k0 + kb + ((e < 8) ? e : e + 8);
                int kc = (k < g.K) ? k : (g.K - 1);
                fa.u[e] = mload(arow + kc, k < g.K);
            }
            const unsigned short* bcol = Bm + (long)(n0 + (lane & 15)) * g.ldbN;
            #pragma unroll
            for (int e = 0; e < 16; e++) {
                int k  = k0 + kb + ((e < 8) ? e : e + 8);
                int kc = (k < g.K) ? k : (g.K - 1);
                fb.u[e] = mload(bcol + (long)kc * g.ldbK, k < g.K);
            }
            acc = wmma_step(fa, fb, acc);
        }
    } else {
        // -------- edge tiles (small N heads/gates): fully branchless --------
        const bool mok = (mA < g.M);
        const unsigned short* arowS = A + (long)(mok ? mA : 0) * g.ldaM;
        const int nlim = g.N - n0;                 // how many of the 16 cols are real
        for (int k0 = 0; k0 < g.K; k0 += 32) {
            Frag16 fa, fb;
            #pragma unroll
            for (int e = 0; e < 16; e++) {
                int k  = k0 + kb + ((e < 8) ? e : e + 8);
                int kc = (k < g.K) ? k : (g.K - 1);
                fa.u[e] = mload(arowS + kc, mok && (k < g.K));
            }
            const bool nok = (lane & 15) < nlim;
            const unsigned short* bcol = Bm + (long)(nok ? n0 + (lane & 15) : n0) * g.ldbN;
            #pragma unroll
            for (int e = 0; e < 16; e++) {
                int k  = k0 + kb + ((e < 8) ? e : e + 8);
                int kc = (k < g.K) ? k : (g.K - 1);
                fb.u[e] = mload(bcol + (long)kc * g.ldbK, nok && (k < g.K));
            }
            acc = wmma_step(fa, fb, acc);
        }
    }

    // ---- epilogue: lane holds col n = n0+(lane&15), rows r + (lane>>4)*8
    float*          Cf = g.Cf ? g.Cf + (long)zo * g.cOuter + (long)zi * g.cInner : nullptr;
    unsigned short* Cb = g.Cb ? g.Cb + (long)zo * g.cOuter + (long)zi * g.cInner : nullptr;
    const float*    R  = g.res ? g.res + (long)zo * g.rOuter + (long)zi * g.rInner : nullptr;
    const int n = n0 + (lane & 15);
    if (n < g.N) {
        const float bs = g.bias ? g.bias[n] : 0.f;
        #pragma unroll
        for (int r = 0; r < 8; r++) {
            int m = m0 + ((lane >> 4) << 3) + r;
            if (m < g.M) {
                float v = acc[r] + bs;
                if (g.act == 1)      v = 0.5f * v * (1.f + erff(v * 0.70710678118f));
                else if (g.act == 2) v = fmaxf(v, 0.f);
                if (R)  v += R[(long)m * g.ldrM + n];
                if (Cf) Cf[(long)m * g.ldcM + n] = v;
                if (Cb) Cb[(long)m * g.ldcM + n] = f2bf(v);
            }
        }
    }
}

// ---------------- support kernels ----------------
__global__ void k_f32bf16(unsigned short* dst, const float* src, long n) {
    long i = (long)blockIdx.x * blockDim.x + threadIdx.x;
    if (i < n) dst[i] = f2bf(src[i]);
}
__global__ void k_fill(float* p, float v, long n) {
    long i = (long)blockIdx.x * blockDim.x + threadIdx.x;
    if (i < n) p[i] = v;
}
__global__ void k_copyf(float* d, const float* s, long n) {
    long i = (long)blockIdx.x * blockDim.x + threadIdx.x;
    if (i < n) d[i] = s[i];
}

__global__ void k_embed(const int* ids, const float* be, const float* bde,
                        const float* cte, const float* cnte, const float* cpe,
                        unsigned short* out) {
    int row = blockIdx.x;
    int id  = ids[row];
    int cont = (id >= 128 && id < 192);
    int bnd  = cont ? 0 : 1;
    int cnt  = 1 + (id >= 192) + (id >= 224) + (id >= 240);
    unsigned short* o = out + (long)row * 1024;
    for (int c = threadIdx.x; c < 1024; c += 256) {
        float v;
        if      (c < 512) v = be[(long)id * 512 + c];
        else if (c < 640) v = bde[bnd * 128 + (c - 512)];
        else if (c < 768) v = cte[c - 640];
        else if (c < 896) v = cnte[cnt * 128 + (c - 768)];
        else              v = cpe[c - 896];
        o[c] = f2bf(v);
    }
}

__global__ void k_posenc(float* x, int d, int S, long n) {
    long i = (long)blockIdx.x * blockDim.x + threadIdx.x;
    if (i >= n) return;
    long t = i / d;
    int  c = (int)(i % d);
    int  s = (int)(t % S);
    float div = expf(-(float)(c & ~1) * (logf(10000.f) / (float)d));
    float ang = (float)s * div;
    x[i] += (c & 1) ? cosf(ang) : sinf(ang);
}

__global__ void k_ln(const float* in, float* out, const float* gg, const float* bb, int d) {
    long row = blockIdx.x;
    const float* x = in + row * d;
    float* o = out + row * d;
    __shared__ float s1[256], s2[256];
    int tid = threadIdx.x;
    float a = 0.f, b = 0.f;
    for (int c = tid; c < d; c += 256) { float v = x[c]; a += v; b += v * v; }
    s1[tid] = a; s2[tid] = b; __syncthreads();
    for (int s = 128; s > 0; s >>= 1) {
        if (tid < s) { s1[tid] += s1[tid + s]; s2[tid] += s2[tid + s]; }
        __syncthreads();
    }
    float mean = s1[0] / d;
    float var  = s2[0] / d - mean * mean;
    float r = rsqrtf(var + 1e-5f);
    for (int c = tid; c < d; c += 256) o[c] = (x[c] - mean) * r * gg[c] + bb[c];
}

// masked, scaled softmax over bf16 score rows (in place)
__global__ void k_softmax(unsigned short* sc, const float* valid, int S, int nh, float scale) {
    int row = blockIdx.x;                // row = (b*nh + h)*S + q
    int b = row / (nh * S);
    unsigned short* p = sc + (long)row * S;
    const float* vb = valid + (long)b * S;
    __shared__ float red[256];
    int tid = threadIdx.x;
    float mx = -3.4e38f;
    for (int k = tid; k < S; k += 256) {
        float v = (vb[k] > 0.5f) ? bf2f(p[k]) * scale : -1e9f;
        mx = fmaxf(mx, v);
    }
    red[tid] = mx; __syncthreads();
    for (int s = 128; s > 0; s >>= 1) { if (tid < s) red[tid] = fmaxf(red[tid], red[tid + s]); __syncthreads(); }
    mx = red[0]; __syncthreads();
    float sum = 0.f;
    for (int k = tid; k < S; k += 256) {
        float v = (vb[k] > 0.5f) ? bf2f(p[k]) * scale : -1e9f;
        sum += expf(v - mx);
    }
    red[tid] = sum; __syncthreads();
    for (int s = 128; s > 0; s >>= 1) { if (tid < s) red[tid] += red[tid + s]; __syncthreads(); }
    sum = red[0]; __syncthreads();
    float inv = 1.f / sum;
    for (int k = tid; k < S; k += 256) {
        float v = (vb[k] > 0.5f) ? bf2f(p[k]) * scale : -1e9f;
        p[k] = f2bf(expf(v - mx) * inv);
    }
}

__global__ void k_pairs(const float* x, unsigned short* out, int d, int S) {
    int row = blockIdx.x;                // row = b*(S-1)+t
    int b = row / (S - 1);
    int t = row % (S - 1);
    const float* x0 = x + ((long)b * S + t) * d;
    unsigned short* o = out + (long)row * 2 * d;
    for (int c = threadIdx.x; c < 2 * d; c += 256) {
        float v = (c < d) ? x0[c] : x0[d + (c - d)];
        o[c] = f2bf(v);
    }
}

__global__ void k_dec(float* dec, const float* glog, const int* ids, const float* valid,
                      int layer, int S, int n, float thr) {
    int i = blockIdx.x * 256 + threadIdx.x;
    if (i >= n) return;
    int b = i / (S - 1), t = i % (S - 1);
    float d;
    if (layer == 0) {
        int id = ids[(long)b * S + t + 1];
        d = (id >= 128 && id < 192) ? 1.f : 0.f;
    } else {
        float p = 1.f / (1.f + expf(-glog[i]));
        d = (p > thr) ? 1.f : 0.f;
    }
    d *= valid[(long)b * S + t + 1];
    dec[i] = d;
}

__global__ void k_scan(const float* dec, const float* valid, int* gid, float* nvalid, int S) {
    int b = blockIdx.x;
    const float* db = dec + (long)b * (S - 1);
    const float* vb = valid + (long)b * S;
    int* gb = gid + (long)b * S;
    float run = 0.f, ng = 1.f;
    gb[0] = 0;
    for (int s = 1; s < S; s++) {
        float d = db[s - 1];
        run += 1.f - d;
        gb[s] = (int)(run + 0.5f);
        ng += (1.f - d) * vb[s];
    }
    float* nv = nvalid + (long)b * S;
    int ngi = (int)(ng + 0.5f);
    for (int s = 0; s < S; s++) nv[s] = (s < ngi) ? 1.f : 0.f;
}

__global__ void k_segmean(const float* x, const int* gid, float* nx, int d, int S, int n) {
    int i = blockIdx.x * 256 + threadIdx.x;
    if (i >= n) return;
    int b = i / d, c = i % d;
    const float* xb = x + (long)b * S * d + c;
    const int*   g  = gid + (long)b * S;
    float* nb = nx + (long)b * S * d + c;
    int cur = 0; float acc = 0.f, cnt = 0.f;
    for (int s = 0; s < S; s++) {
        int gs = g[s];
        if (gs != cur) {
            nb[(long)cur * d] = acc / fmaxf(cnt, 1.f);
            cur = gs; acc = 0.f; cnt = 0.f;
        }
        acc += xb[(long)s * d];
        cnt += 1.f;
    }
    nb[(long)cur * d] = acc / fmaxf(cnt, 1.f);
}

__global__ void k_amean(const float* att, const float* valid, float* am, int d, int S, int n) {
    int i = blockIdx.x * 256 + threadIdx.x;
    if (i >= n) return;
    int b = i / d, c = i % d;
    const float* a = att + (long)b * S * d + c;
    const float* v = valid + (long)b * S;
    float s = 0.f, cv = 0.f;
    for (int t = 0; t < S; t++) { s += a[(long)t * d] * v[t]; cv += v[t]; }
    am[i] = s / fmaxf(cv, 1.f);
}

__global__ void k_combine(float* x, const float* nx, const float* am, const float* nv,
                          int d, int S, long n) {
    long i = (long)blockIdx.x * blockDim.x + threadIdx.x;
    if (i >= n) return;
    long bs = i / d;
    int  c  = (int)(i % d);
    int  b  = (int)(bs / S);
    int  s  = (int)(bs % S);
    x[i] = (nx[i] + 0.1f * am[(long)b * d + c]) * nv[(long)b * S + s];
}

// ---------------- host orchestration ----------------
extern "C" void kernel_launch(void* const* d_in, const int* in_sizes, int n_in,
                              void* d_out, int out_size, void* d_ws, size_t ws_size,
                              hipStream_t stream) {
    (void)in_sizes; (void)n_in; (void)out_size; (void)ws_size;
    constexpr int  Bb = 16, Ss = 512;
    constexpr long NT = (long)Bb * Ss;
    const int HDv[5] = {512, 512, 640, 768, 768};

    int cur = 0;
    const int* ids = (const int*)d_in[cur++];
    auto F = [&]() { return (const float*)d_in[cur++]; };
    const float* byte_emb = F();  const float* bnd_emb = F();
    const float* ct_emb   = F();  const float* cnt_emb = F();  const float* cp_emb = F();
    const float* inpw = F();      const float* inpb = F();
    struct LayerP {
        const float *projw, *projb, *ln1g, *ln1b, *wqkv, *bqkv, *wo, *bo,
                    *ln2g, *ln2b, *ff1w, *ff1b, *ff2w, *ff2b, *ng, *nb,
                    *g1w, *g1b, *g2w, *g2b, *mqkvw, *mqkvb, *mwow, *mwob;
    } LP[5];
    for (int i = 0; i < 5; i++) {
        LayerP& L = LP[i];
        if (i == 2 || i == 3) { L.projw = F(); L.projb = F(); }
        else { L.projw = nullptr; L.projb = nullptr; }
        L.ln1g = F(); L.ln1b = F();
        L.wqkv = F(); L.bqkv = F(); L.wo = F(); L.bo = F();
        L.ln2g = F(); L.ln2b = F();
        L.ff1w = F(); L.ff1b = F(); L.ff2w = F(); L.ff2b = F();
        L.ng = F();   L.nb = F();
        L.g1w = F();  L.g1b = F();  L.g2w = F();  L.g2b = F();
        L.mqkvw = F(); L.mqkvb = F(); L.mwow = F(); L.mwob = F();
    }
    const float* bpw = F(); const float* bpb = F();
    const float* cpw = F(); const float* cpb = F();

    // ---- workspace carve ----
    char* base = (char*)d_ws;
    size_t off = 0;
    auto alc = [&](size_t bytes) { size_t o = off; off += (bytes + 255) & ~(size_t)255; return o; };
    float* X     = (float*)(base + alc(NT * 768 * 4));
    float* Y     = (float*)(base + alc(NT * 768 * 4));
    float* HX    = (float*)(base + alc(NT * 768 * 4));
    float* T     = (float*)(base + alc(NT * 768 * 4));
    float* ATT   = (float*)(base + alc(NT * 768 * 4));
    float* NEWX  = (float*)(base + alc(NT * 768 * 4));
    float* VALID = (float*)(base + alc(NT * 4));
    float* NVAL  = (float*)(base + alc(NT * 4));
    float* DEC   = (float*)(base + alc((size_t)Bb * (Ss - 1) * 4));
    float* GLOG  = (float*)(base + alc((size_t)Bb * (Ss - 1) * 4));
    float* AMEAN = (float*)(base + alc((size_t)Bb * 768 * 4));
    int*   GID   = (int*)  (base + alc(NT * 4));
    unsigned short* ABF   = (unsigned short*)(base + alc(NT * 2048 * 2));
    unsigned short* FBF   = (unsigned short*)(base + alc(NT * 3072 * 2));
    unsigned short* QKVBF = (unsigned short*)(base + alc(NT * 2304 * 2));
    unsigned short* SCBF  = (unsigned short*)(base + alc((size_t)Bb * 8 * Ss * Ss * 2));
    unsigned short* WBF   = (unsigned short*)(base + alc((size_t)3072 * 768 * 2));

    auto blocksOf = [](long n) { return (unsigned)((n + 255) / 256); };
    auto toBF = [&](unsigned short* dst, const float* src, long n) {
        k_f32bf16<<<blocksOf(n), 256, 0, stream>>>(dst, src, n);
    };
    auto GEMM = [&](const unsigned short* A, long aO, long aI, long ldaM,
                    const unsigned short* Bw, long bO, long bI, long ldbK, long ldbN,
                    float* Cf, unsigned short* Cb, long cO, long cI, long ldcM,
                    const float* bias, const float* res, long rO, long rI, long ldrM,
                    int M, int N, int K, int act, int batch, int bdiv) {
        GemmArgs g;
        g.A = A; g.B = Bw; g.Cf = Cf; g.Cb = Cb; g.bias = bias; g.res = res;
        g.aOuter = aO; g.aInner = aI; g.ldaM = ldaM;
        g.bOuter = bO; g.bInner = bI; g.ldbK = ldbK; g.ldbN = ldbN;
        g.cOuter = cO; g.cInner = cI; g.ldcM = ldcM;
        g.rOuter = rO; g.rInner = rI; g.ldrM = ldrM;
        g.M = M; g.N = N; g.K = K; g.act = act;
        g.batchDiv = (bdiv < 1) ? 1 : bdiv;
        g.tilesM32 = (M + 31) / 32;
        int tN = (N + 31) / 32;
        dim3 grid((unsigned)(g.tilesM32 * tN), (unsigned)batch);
        k_gemm<<<grid, 128, 0, stream>>>(g);
    };

    // ---- stem ----
    k_fill<<<blocksOf(NT), 256, 0, stream>>>(VALID, 1.f, NT);
    k_embed<<<(unsigned)NT, 256, 0, stream>>>(ids, byte_emb, bnd_emb, ct_emb, cnt_emb, cp_emb, ABF);
    toBF(WBF, inpw, 1024L * 512);
    GEMM(ABF, 0, 0, 1024, WBF, 0, 0, 512, 1, X, nullptr, 0, 0, 512,
         inpb, nullptr, 0, 0, 0, (int)NT, 512, 1024, 0, 1, 1);
    k_posenc<<<blocksOf(NT * 512), 256, 0, stream>>>(X, 512, Ss, NT * 512);

    int din = 512;
    for (int i = 0; i < 5; i++) {
        const int d = HDv[i];
        const LayerP& L = LP[i];
        if (L.projw) {
            toBF(ABF, X, NT * din);
            toBF(WBF, L.projw, (long)din * d);
            GEMM(ABF, 0, 0, din, WBF, 0, 0, d, 1, X, nullptr, 0, 0, d,
                 L.projb, nullptr, 0, 0, 0, (int)NT, d, din, 0, 1, 1);
        }
        // ---- encoder layer ----
        const int dh = d / 8;
        k_ln<<<(unsigned)NT, 256, 0, stream>>>(X, Y, L.ln1g, L.ln1b, d);
        toBF(ABF, Y, NT * d);
        toBF(WBF, L.wqkv, (long)d * 3 * d);
        GEMM(ABF, 0, 0, d, WBF, 0, 0, 3 * d, 1, nullptr, QKVBF, 0, 0, 3 * d,
             L.bqkv, nullptr, 0, 0, 0, (int)NT, 3 * d, d, 0, 1, 1);
        // scores = Q @ K^T  (bf16 out)
        GEMM(QKVBF, (long)Ss * 3 * d, dh, 3 * d,
             QKVBF + d, (long)Ss * 3 * d, dh, 1, 3 * d,
             nullptr, SCBF, (long)8 * Ss * Ss, (long)Ss * Ss, Ss,
             nullptr, nullptr, 0, 0, 0, Ss, Ss, dh, 0, Bb * 8, 8);
        k_softmax<<<(unsigned)(Bb * 8 * Ss), 256, 0, stream>>>(SCBF, VALID, Ss, 8, 1.f / sqrtf((float)dh));
        // O = P @ V  (bf16 out -> ABF)
        GEMM(SCBF, (long)8 * Ss * Ss, (long)Ss * Ss, Ss,
             QKVBF + 2 * d, (long)Ss * 3 * d, dh, 3 * d, 1,
             nullptr, ABF, (long)Ss * d, dh, d,
             nullptr, nullptr, 0, 0, 0, Ss, dh, Ss, 0, Bb * 8, 8);
        toBF(WBF, L.wo, (long)d * d);
        GEMM(ABF, 0, 0, d, WBF, 0, 0, d, 1, HX, nullptr, 0, 0, d,
             L.bo, X, 0, 0, d, (int)NT, d, d, 0, 1, 1);
        k_ln<<<(unsigned)NT, 256, 0, stream>>>(HX, Y, L.ln2g, L.ln2b, d);
        toBF(ABF, Y, NT * d);
        toBF(WBF, L.ff1w, (long)d * 4 * d);
        GEMM(ABF, 0, 0, d, WBF, 0, 0, 4 * d, 1, nullptr, FBF, 0, 0, 4 * d,
             L.ff1b, nullptr, 0, 0, 0, (int)NT, 4 * d, d, 1, 1, 1);   // fused exact GELU
        toBF(WBF, L.ff2w, (long)4 * d * d);
        GEMM(FBF, 0, 0, 4 * d, WBF, 0, 0, d, 1, T, nullptr, 0, 0, d,
             L.ff2b, HX, 0, 0, d, (int)NT, d, 4 * d, 0, 1, 1);
        k_ln<<<(unsigned)NT, 256, 0, stream>>>(T, X, L.ng, L.nb, d);

        // ---- merge attention (MH = 4) ----
        const int mdh = d / 4;
        toBF(ABF, X, NT * d);
        toBF(WBF, L.mqkvw, (long)d * 3 * d);
        GEMM(ABF, 0, 0, d, WBF, 0, 0, 3 * d, 1, nullptr, QKVBF, 0, 0, 3 * d,
             L.mqkvb, nullptr, 0, 0, 0, (int)NT, 3 * d, d, 0, 1, 1);
        GEMM(QKVBF, (long)Ss * 3 * d, mdh, 3 * d,
             QKVBF + d, (long)Ss * 3 * d, mdh, 1, 3 * d,
             nullptr, SCBF, (long)4 * Ss * Ss, (long)Ss * Ss, Ss,
             nullptr, nullptr, 0, 0, 0, Ss, Ss, mdh, 0, Bb * 4, 4);
        k_softmax<<<(unsigned)(Bb * 4 * Ss), 256, 0, stream>>>(SCBF, VALID, Ss, 4, 1.f / sqrtf((float)mdh));
        GEMM(SCBF, (long)4 * Ss * Ss, (long)Ss * Ss, Ss,
             QKVBF + 2 * d, (long)Ss * 3 * d, mdh, 3 * d, 1,
             nullptr, ABF, (long)Ss * d, mdh, d,
             nullptr, nullptr, 0, 0, 0, Ss, mdh, Ss, 0, Bb * 4, 4);
        toBF(WBF, L.mwow, (long)d * d);
        GEMM(ABF, 0, 0, d, WBF, 0, 0, d, 1, ATT, nullptr, 0, 0, d,
             L.mwob, nullptr, 0, 0, 0, (int)NT, d, d, 0, 1, 1);
        k_amean<<<blocksOf((long)Bb * d), 256, 0, stream>>>(ATT, VALID, AMEAN, d, Ss, Bb * d);

        // ---- boundary gates ----
        if (i > 0) {
            k_pairs<<<(unsigned)(Bb * (Ss - 1)), 256, 0, stream>>>(X, ABF, d, Ss);
            toBF(WBF, L.g1w, (long)2 * d * d);
            GEMM(ABF, 0, 0, 2 * d, WBF, 0, 0, d, 1, nullptr, FBF, 0, 0, d,
                 L.g1b, nullptr, 0, 0, 0, Bb * (Ss - 1), d, 2 * d, 2, 1, 1); // fused ReLU
            toBF(WBF, L.g2w, (long)d);
            GEMM(FBF, 0, 0, d, WBF, 0, 0, 1, 1, GLOG, nullptr, 0, 0, 1,
                 L.g2b, nullptr, 0, 0, 0, Bb * (Ss - 1), 1, d, 0, 1, 1);
        }
        const float thr = 0.7f + (float)i / 5.f * 0.2f;
        k_dec<<<blocksOf((long)Bb * (Ss - 1)), 256, 0, stream>>>(DEC, GLOG, ids, VALID, i, Ss, Bb * (Ss - 1), thr);
        k_scan<<<(unsigned)Bb, 1, 0, stream>>>(DEC, VALID, GID, NVAL, Ss);
        k_fill<<<blocksOf(NT * d), 256, 0, stream>>>(NEWX, 0.f, NT * d);
        k_segmean<<<blocksOf((long)Bb * d), 256, 0, stream>>>(X, GID, NEWX, d, Ss, Bb * d);
        k_combine<<<blocksOf(NT * d), 256, 0, stream>>>(X, NEWX, AMEAN, NVAL, d, Ss, NT * d);
        k_copyf<<<blocksOf(NT), 256, 0, stream>>>(VALID, NVAL, NT);
        din = d;
    }

    // ---- heads + output assembly: [x | bl | cl | valid] ----
    float* out = (float*)d_out;
    k_copyf<<<blocksOf(NT * 768), 256, 0, stream>>>(out, X, NT * 768);
    toBF(ABF, X, NT * 768);
    toBF(WBF, bpw, 768L * 4);
    GEMM(ABF, 0, 0, 768, WBF, 0, 0, 4, 1, out + NT * 768, nullptr, 0, 0, 4,
         bpb, nullptr, 0, 0, 0, (int)NT, 4, 768, 0, 1, 1);
    toBF(WBF, cpw, 768L * 14);
    GEMM(ABF, 0, 0, 768, WBF, 0, 0, 14, 1, out + NT * 772, nullptr, 0, 0, 14,
         cpb, nullptr, 0, 0, 0, (int)NT, 14, 768, 0, 1, 1);
    k_copyf<<<blocksOf(NT), 256, 0, stream>>>(out + NT * 786, VALID, NT);
}
